// Attention_18528488915211
// MI455X (gfx1250) — matrix operand order repaired and checked by reading, not compiled, not run on hardware
//
#include <hip/hip_runtime.h>
#include <stdint.h>

// ---------------- problem constants ----------------
#define BATCH   4
#define SEQ     2048
#define DMODEL  2048
#define HEADS   16
#define DHEAD   128           // DMODEL / HEADS
#define MROWS   (BATCH * SEQ) // 8192

// ---------------- WMMA types ----------------
typedef __attribute__((ext_vector_type(16))) __bf16 v16bf;
typedef __attribute__((ext_vector_type(8)))  float  v8f;

union V16U {
    v16bf v;
    uint4 q[2];
};

static __device__ __forceinline__ v8f wmma_bf16(v16bf a, v16bf b, v8f c) {
    return __builtin_amdgcn_wmma_f32_16x16x32_bf16(
        /*neg_a=*/false, a, /*neg_b=*/false, b,
        /*c_mod=*/(short)0, c, /*reuse_a=*/false, /*reuse_b=*/false);
}

// LDS byte address of a shared-memory pointer (as(3) pointers are 32-bit offsets)
static __device__ __forceinline__ uint32_t lds_addr(const void* p) {
    return (uint32_t)(size_t)(__attribute__((address_space(3))) const void*)p;
}

// CDNA5 async copy: global -> LDS, 16B per lane, tracked by ASYNCcnt (ISA 10.7/15.18)
static __device__ __forceinline__ void async_load_b128(uint32_t lds, uint64_t gaddr) {
    asm volatile("global_load_async_to_lds_b128 %0, %1, off"
                 :: "v"(lds), "v"(gaddr) : "memory");
}

// =====================================================================
// Prep kernel A: x (f32) -> xb (bf16), 8 elements / thread, b128 in/out
// =====================================================================
__global__ __launch_bounds__(256) void convert_x_kernel(
    const float* __restrict__ X, __bf16* __restrict__ Xb)
{
    size_t idx = ((size_t)blockIdx.x * 256 + threadIdx.x) * 8;
    float4 f0 = *(const float4*)(X + idx);
    float4 f1 = *(const float4*)(X + idx + 4);
    union { __bf16 h[8]; uint4 q; } o;
    o.h[0] = (__bf16)f0.x; o.h[1] = (__bf16)f0.y;
    o.h[2] = (__bf16)f0.z; o.h[3] = (__bf16)f0.w;
    o.h[4] = (__bf16)f1.x; o.h[5] = (__bf16)f1.y;
    o.h[6] = (__bf16)f1.z; o.h[7] = (__bf16)f1.w;
    *(uint4*)(Xb + idx) = o.q;
}

// =====================================================================
// Prep kernel B: Wt[n][k] = bf16(W[k][n]).  64x64 tiles through LDS so
// both the global read (over n) and write (over k) are coalesced.
// =====================================================================
__global__ __launch_bounds__(256) void transpose_w_kernel(
    const float* __restrict__ W, __bf16* __restrict__ Wt)
{
    __shared__ __attribute__((aligned(16))) __bf16 T[64][72]; // padded rows
    const int tid = threadIdx.x;
    const int n0 = blockIdx.x * 64;
    const int k0 = blockIdx.y * 64;
#pragma unroll
    for (int i = 0; i < 16; ++i) {
        int linear = i * 256 + tid;          // 0..4095
        int k = linear >> 6, n = linear & 63;
        T[n][k] = (__bf16)W[(size_t)(k0 + k) * DMODEL + (n0 + n)];
    }
    __syncthreads();
#pragma unroll
    for (int i = 0; i < 4; ++i) {
        int linear = i * 256 + tid;          // 0..1023
        int n = linear >> 4, kg = (linear & 15) * 4;
        *(uint2*)&Wt[(size_t)(n0 + n) * DMODEL + (k0 + kg)] = *(const uint2*)&T[n][kg];
    }
}

// =====================================================================
// Unified bf16 GEMM: C = A[M][K] @ Bt[N][K]^T + bias.
//   MODE 0: out f32 row-major (final projection)
//   MODE 1: out bf16 [B,H,L,DHEAD]   (Q, K)
//   MODE 2: out bf16 [B,H,DHEAD,L]   (V, pre-transposed for attention)
// Block tile 128x128, 8 waves of 32x64, K staged 64 wide via async
// global->LDS copies, double buffered; last iteration peeled so the
// steady-state loop is branch-free.
// =====================================================================
template <int MODE>
__global__ __launch_bounds__(256) void gemm_bf16_kernel(
    const __bf16* __restrict__ A, const __bf16* __restrict__ Bt,
    const float* __restrict__ bias, float* __restrict__ outF,
    __bf16* __restrict__ outB)
{
    __shared__ __attribute__((aligned(16))) __bf16 sm[2][2][128 * 64]; // 64 KB

    const int tid  = threadIdx.x;
    const int lane = tid & 31;
    const int wave = tid >> 5;
    const int hi   = lane >> 4;
    const int l16  = lane & 15;

    const int m0 = blockIdx.x * 128;
    const int n0 = blockIdx.y * 128;
    const int wm = wave & 3;      // 4 M-groups of 32 rows
    const int wn = wave >> 2;     // 2 N-groups of 64 cols

    v8f acc[2][4];
#pragma unroll
    for (int i = 0; i < 2; ++i)
#pragma unroll
        for (int j = 0; j < 4; ++j)
            acc[i][j] = (v8f){0.f,0.f,0.f,0.f,0.f,0.f,0.f,0.f};

    // Issue one 128x64 A tile + one 128x64 B tile as 8 async b128s / thread
    auto issue_tile = [&](int buf, int kc) {
#pragma unroll
        for (int u4 = 0; u4 < 4; ++u4) {
            int u = u4 * 256 + tid;           // 0..1023 16B units per matrix
            int r = u >> 3, c = (u & 7) * 8;  // row, col (halfs)
            async_load_b128(lds_addr(&sm[buf][0][r * 64 + c]),
                            (uint64_t)(size_t)(A  + (size_t)(m0 + r) * DMODEL + kc + c));
            async_load_b128(lds_addr(&sm[buf][1][r * 64 + c]),
                            (uint64_t)(size_t)(Bt + (size_t)(n0 + r) * DMODEL + kc + c));
        }
    };

    // 16 WMMAs on one staged 128x64 double-tile
    auto compute_tile = [&](int buf) {
        const __bf16* Asb = &sm[buf][0][0];
        const __bf16* Bsb = &sm[buf][1][0];
#pragma unroll
        for (int ks = 0; ks < 2; ++ks) {
            v16bf a[2];
#pragma unroll
            for (int i = 0; i < 2; ++i) {
                int m = wm * 32 + i * 16 + l16;
                V16U u;
                u.q[0] = *(const uint4*)&Asb[m * 64 + ks * 32 + hi * 8];
                u.q[1] = *(const uint4*)&Asb[m * 64 + ks * 32 + 16 + hi * 8];
                a[i] = u.v;
            }
            v16bf b[4];
#pragma unroll
            for (int j = 0; j < 4; ++j) {
                int n = wn * 64 + j * 16 + l16;
                V16U u;
                u.q[0] = *(const uint4*)&Bsb[n * 64 + ks * 32 + hi * 8];
                u.q[1] = *(const uint4*)&Bsb[n * 64 + ks * 32 + 16 + hi * 8];
                b[j] = u.v;
            }
#pragma unroll
            for (int i = 0; i < 2; ++i)
#pragma unroll
                for (int j = 0; j < 4; ++j)
                    acc[i][j] = wmma_bf16(a[i], b[j], acc[i][j]);
        }
    };

    issue_tile(0, 0);
    // Steady state: branch-free; prefetch of tile k+1 overlaps compute of tile k
#pragma unroll 1
    for (int kc = 0; kc < DMODEL - 64; kc += 64) {
        const int buf = (kc >> 6) & 1;
        issue_tile(buf ^ 1, kc + 64);
        asm volatile("s_wait_asynccnt 8" ::: "memory");   // oldest 8 (current tile) done
        __syncthreads();
        compute_tile(buf);
        __syncthreads();   // all reads done before this buffer is refilled
    }
    // Peeled tail
    asm volatile("s_wait_asynccnt 0" ::: "memory");
    __syncthreads();
    compute_tile(((DMODEL - 64) >> 6) & 1);

    // ---------------- epilogue (compile-time specialized) ----------------
#pragma unroll
    for (int j = 0; j < 4; ++j) {
        int col = n0 + wn * 64 + j * 16 + l16;
        float bv = bias[col];
        int h = col >> 7, d = col & (DHEAD - 1);
#pragma unroll
        for (int i = 0; i < 2; ++i) {
#pragma unroll
            for (int r = 0; r < 8; ++r) {
                int m  = m0 + wm * 32 + i * 16 + r + 8 * hi;
                float val = acc[i][j][r] + bv;
                if (MODE == 0) {
                    outF[(size_t)m * DMODEL + col] = val;
                } else if (MODE == 1) {
                    int bb = m >> 11, ll = m & (SEQ - 1);
                    outB[(((size_t)(bb * HEADS + h)) * SEQ + ll) * DHEAD + d] = (__bf16)val;
                } else {
                    int bb = m >> 11, ll = m & (SEQ - 1);
                    outB[(((size_t)(bb * HEADS + h)) * DHEAD + d) * SEQ + ll] = (__bf16)val;
                }
            }
        }
    }
}

// =====================================================================
// Flash attention.  Q,K: [B,H,L,D]; V: [B,H,D,L] (pre-transposed).
// Each wave owns 16 query rows; waves fully independent (no barriers).
// =====================================================================
__global__ __launch_bounds__(256) void attn_kernel(
    const __bf16* __restrict__ Qp, const __bf16* __restrict__ Kp,
    const __bf16* __restrict__ Vt, __bf16* __restrict__ PF)
{
    __shared__ __attribute__((aligned(16))) __bf16 Psh[8][16][32];  // per-wave P transpose

    const int tid  = threadIdx.x;
    const int lane = tid & 31;
    const int wave = tid >> 5;
    const int hi   = lane >> 4;
    const int l16  = lane & 15;

    const int bh = blockIdx.y;          // b*16 + h
    const int b  = bh >> 4;
    const int h  = bh & 15;
    const size_t hbase = (size_t)bh * SEQ * DHEAD;
    const int q0 = blockIdx.x * 128 + wave * 16;

    // Preload Q fragments (16 rows x 128 d) as 4 A-frags
    v16bf qf[4];
#pragma unroll
    for (int dc = 0; dc < 4; ++dc) {
        const __bf16* qrow = Qp + hbase + (size_t)(q0 + l16) * DHEAD + dc * 32;
        V16U u;
        u.q[0] = *(const uint4*)(qrow + hi * 8);
        u.q[1] = *(const uint4*)(qrow + 16 + hi * 8);
        qf[dc] = u.v;
    }

    v8f o[8];
#pragma unroll
    for (int dt = 0; dt < 8; ++dt)
        o[dt] = (v8f){0.f,0.f,0.f,0.f,0.f,0.f,0.f,0.f};
    float mrow[8], lrow[8];
#pragma unroll
    for (int r = 0; r < 8; ++r) { mrow[r] = -3.0e38f; lrow[r] = 0.0f; }

    const float sscale = 1.0f / (float)DHEAD;   // reference applies 1/sqrt(d) twice

#pragma unroll 1
    for (int kv = 0; kv < SEQ; kv += 32) {
        // S = Q K^T for 32 kv columns (two 16x16 tiles); K rows contiguous in d
        v8f s[2];
#pragma unroll
        for (int t = 0; t < 2; ++t) {
            v8f sacc = (v8f){0.f,0.f,0.f,0.f,0.f,0.f,0.f,0.f};
            const __bf16* krow = Kp + hbase + (size_t)(kv + t * 16 + l16) * DHEAD;
#pragma unroll
            for (int dc = 0; dc < 4; ++dc) {
                V16U u;
                u.q[0] = *(const uint4*)(krow + dc * 32 + hi * 8);
                u.q[1] = *(const uint4*)(krow + dc * 32 + 16 + hi * 8);
                sacc = wmma_bf16(qf[dc], u.v, sacc);
            }
            s[t] = sacc;
        }

        // Online softmax: row stats via 16-lane half-group shuffles
#pragma unroll
        for (int r = 0; r < 8; ++r) {
            float s0 = s[0][r] * sscale;
            float s1 = s[1][r] * sscale;
            float mx = fmaxf(s0, s1);
#pragma unroll
            for (int off = 8; off >= 1; off >>= 1)
                mx = fmaxf(mx, __shfl_xor(mx, off, 16));
            float mnew = fmaxf(mrow[r], mx);
            float corr = __expf(mrow[r] - mnew);
            float p0 = __expf(s0 - mnew);
            float p1 = __expf(s1 - mnew);
            float rs = p0 + p1;
#pragma unroll
            for (int off = 8; off >= 1; off >>= 1)
                rs += __shfl_xor(rs, off, 16);
            lrow[r] = lrow[r] * corr + rs;
            mrow[r] = mnew;
            Psh[wave][r + 8 * hi][l16]      = (__bf16)p0;
            Psh[wave][r + 8 * hi][16 + l16] = (__bf16)p1;
#pragma unroll
            for (int dt = 0; dt < 8; ++dt) o[dt][r] *= corr;
        }
        // wave-private LDS RAW: drain DS counter before re-reading P
        asm volatile("s_wait_dscnt 0" ::: "memory");

        // Reload P in A-fragment layout
        V16U up;
        up.q[0] = *(const uint4*)&Psh[wave][l16][hi * 8];
        up.q[1] = *(const uint4*)&Psh[wave][l16][16 + hi * 8];
        v16bf pfrag = up.v;

        // O += P (16x32) x V (32x128); V pre-transposed: contiguous in kv
#pragma unroll
        for (int dt = 0; dt < 8; ++dt) {
            const __bf16* vcol = Vt + hbase + (size_t)(dt * 16 + l16) * SEQ + kv + hi * 8;
            V16U uv;
            uv.q[0] = *(const uint4*)vcol;
            uv.q[1] = *(const uint4*)(vcol + 16);
            o[dt] = wmma_bf16(pfrag, uv.v, o[dt]);
        }
    }

    // Normalize and store prefinal as bf16 [b, l, h*DHEAD + d]
#pragma unroll
    for (int r = 0; r < 8; ++r) {
        float inv = 1.0f / lrow[r];
        int m = q0 + r + 8 * hi;
        size_t ob = ((size_t)b * SEQ + m) * DMODEL + h * DHEAD;
#pragma unroll
        for (int dt = 0; dt < 8; ++dt)
            PF[ob + dt * 16 + l16] = (__bf16)(o[dt][r] * inv);
    }
}

// =====================================================================
// Host-side launcher
// =====================================================================
extern "C" void kernel_launch(void* const* d_in, const int* in_sizes, int n_in,
                              void* d_out, int out_size, void* d_ws, size_t ws_size,
                              hipStream_t stream) {
    const float* x  = (const float*)d_in[0];
    const float* Wq = (const float*)d_in[1];
    const float* bq = (const float*)d_in[2];
    const float* Wk = (const float*)d_in[3];
    const float* bk = (const float*)d_in[4];
    const float* Wv = (const float*)d_in[5];
    const float* bv = (const float*)d_in[6];
    const float* Wo = (const float*)d_in[7];
    const float* bo = (const float*)d_in[8];

    const size_t XE = (size_t)MROWS * DMODEL;        // 16.7M halfs
    const size_t WE = (size_t)DMODEL * DMODEL;       // 4.2M halfs
    __bf16* Xb  = (__bf16*)d_ws;
    __bf16* WtQ = Xb  + XE;
    __bf16* WtK = WtQ + WE;
    __bf16* WtV = WtK + WE;
    __bf16* WtO = WtV + WE;
    __bf16* Qp  = WtO + WE;
    __bf16* Kp  = Qp  + XE;
    __bf16* Vt  = Kp  + XE;
    __bf16* PF  = Vt  + XE;
    if (ws_size < (5 * XE + 4 * WE) * sizeof(__bf16)) return;

    dim3 blk(256);
    dim3 gCvt(XE / (256 * 8));                   // 8192
    dim3 gTrp(DMODEL / 64, DMODEL / 64);         // 32 x 32
    dim3 gGemm(MROWS / 128, DMODEL / 128);       // 64 x 16
    dim3 gAttn(SEQ / 128, BATCH * HEADS);        // 16 x 64

    convert_x_kernel<<<gCvt, blk, 0, stream>>>(x, Xb);
    transpose_w_kernel<<<gTrp, blk, 0, stream>>>(Wq, WtQ);
    transpose_w_kernel<<<gTrp, blk, 0, stream>>>(Wk, WtK);
    transpose_w_kernel<<<gTrp, blk, 0, stream>>>(Wv, WtV);
    transpose_w_kernel<<<gTrp, blk, 0, stream>>>(Wo, WtO);

    gemm_bf16_kernel<1><<<gGemm, blk, 0, stream>>>(Xb, WtQ, bq, nullptr, Qp);
    gemm_bf16_kernel<1><<<gGemm, blk, 0, stream>>>(Xb, WtK, bk, nullptr, Kp);
    gemm_bf16_kernel<2><<<gGemm, blk, 0, stream>>>(Xb, WtV, bv, nullptr, Vt);
    attn_kernel<<<gAttn, blk, 0, stream>>>(Qp, Kp, Vt, PF);
    gemm_bf16_kernel<0><<<gGemm, blk, 0, stream>>>(PF, WtO, bo, (float*)d_out, nullptr);
}
